// Logic_Model_80607946211458
// MI455X (gfx1250) — compile-verified, matrix operand order checked by reading, not run on hardware
//
#include <hip/hip_runtime.h>
#include <hip/hip_bf16.h>
#include <math.h>

// ---- problem constants (mirror the reference) ----
#define NB     500      // batch rows
#define NBODY  30       // body predicates
#define NCOL   32       // body + 2 empty predicates
#define KSEL   3        // top-k per formula
#define SIGMA  0.1f
#define TEMP   0.07f
#define TOLR   0.02f
#define NF     2        // formulas (num_formula - 1)

#define ROWS_PAD   512  // 32 tiles of 16 rows
#define WAVES_PER_BLOCK 8
#define THREADS    (WAVES_PER_BLOCK * 32)
#define NBLOCKS    (ROWS_PAD / 16 / WAVES_PER_BLOCK)   // 4

typedef __attribute__((ext_vector_type(16))) _Float16 v16h;
typedef __attribute__((ext_vector_type(8)))  float    v8f;

// K index for half-element e (0..15) of a 16-bit 16x32 WMMA operand, lane L.
// (ISA 7.12.2: lanes 0-15 VGPR0..3 -> K 0..7, VGPR4..7 -> K 16..23;
//  lanes 16-31 VGPR0..3 -> K 8..15, VGPR4..7 -> K 24..31)
__device__ __forceinline__ int wmma_k_of(int lane, int e) {
    int vg = e >> 1, sub = e & 1;
    return ((vg & 3) << 1) + sub + (lane >= 16 ? 8 : 0) + (vg >= 4 ? 16 : 0);
}

__global__ __launch_bounds__(THREADS)
void logic_model_fused(const float* __restrict__ t,          // (500,1)
                       const float* __restrict__ ds,         // (500,30)
                       const float* __restrict__ pi,         // (3,)
                       const float* __restrict__ A,          // (2,32)
                       const float* __restrict__ base_p,     // (1,)
                       const float* __restrict__ fw,         // (2,)
                       const float* __restrict__ prob,       // (4,)
                       float* __restrict__ out)              // (500,3)
{
    __shared__ float sdot[WAVES_PER_BLOCK][16][2];   // per-wave 16 rows x 2 formula dots

    const int lane = threadIdx.x & 31;
    const int wv   = threadIdx.x >> 5;
    const int gw   = blockIdx.x * WAVES_PER_BLOCK + wv;  // global wave = tile id
    const int row0 = gw * 16;

    // ---------------- build A operand: 16x32 indicator tile (f16, exact) ----------------
    const int m  = lane & 15;
    int b        = row0 + m;
    const bool rv = (b < NB);
    const int bc = rv ? b : (NB - 1);
    const float tb_a = t[bc];

    v16h aop;
    #pragma unroll
    for (int e = 0; e < 16; ++e) {
        int k = wmma_k_of(lane, e);
        float indv;
        if (k < NBODY) indv = (ds[bc * NBODY + k] <= tb_a) ? 1.0f : 0.0f;
        else           indv = 1.0f;                      // empty predicates always on
        if (!rv)       indv = 0.0f;                      // padded rows contribute nothing
        aop[e] = (_Float16)indv;
    }

    // ---------------- build B operand: A^T padded to 32x16, split hi/lo f16 ----------------
    const int n = lane & 15;     // output column (formula index; >=2 are zero pads)
    v16h bhi, blo;
    #pragma unroll
    for (int e = 0; e < 16; ++e) {
        int k = wmma_k_of(lane, e);
        float av = (n < NF) ? A[n * NCOL + k] : 0.0f;
        _Float16 hi = (_Float16)av;
        _Float16 lo = (_Float16)(av - (float)hi);
        bhi[e] = hi;
        blo[e] = lo;
    }

    // ---------------- WMMA: D = ind * (B_hi + B_lo), f32 accumulate ----------------
    v8f acc = {};
    acc = __builtin_amdgcn_wmma_f32_16x16x32_f16(false, aop, false, bhi,
                                                 (short)0, acc, false, false);
    acc = __builtin_amdgcn_wmma_f32_16x16x32_f16(false, aop, false, blo,
                                                 (short)0, acc, false, false);

    // scatter the two useful columns to LDS: VGPR r, lane L -> (M = r + (L<16?0:8), N = L%16)
    if (n < NF) {
        int mbase = (lane < 16) ? 0 : 8;
        #pragma unroll
        for (int r = 0; r < 8; ++r) sdot[wv][mbase + r][n] = acc[r];
    }
    __syncthreads();

    // ---------------- scalar epilogue: 16 lanes, one row each ----------------
    if (lane < 16) {
        int brow = row0 + lane;
        if (brow < NB) {
            const float tb   = t[brow];
            const float basv = base_p[0];
            const float p0 = prob[0], p1 = prob[1], p2 = prob[2], p3 = prob[3];

            float dsr[NBODY];
            #pragma unroll
            for (int p = 0; p < NBODY; ++p) dsr[p] = ds[brow * NBODY + p];

            // z = 0 component
            out[brow * 3 + 0] = logf(basv) - tb * basv + logf(pi[0]);

            const int PA[3] = {0, 0, 1};
            const int PB[3] = {1, 2, 2};

            #pragma unroll
            for (int i = 0; i < NF; ++i) {
                // ---- top-3 of A[i] (strict >: first occurrence wins), then sort ascending ----
                int i0 = 0, i1 = 0, i2 = 0;
                float v0 = -1e30f, v1 = -1e30f, v2 = -1e30f;
                for (int k = 0; k < NCOL; ++k) {
                    float v = A[i * NCOL + k];
                    if (v > v0)      { v2 = v1; i2 = i1; v1 = v0; i1 = i0; v0 = v; i0 = k; }
                    else if (v > v1) { v2 = v1; i2 = i1; v1 = v;  i1 = k; }
                    else if (v > v2) { v2 = v;  i2 = k; }
                }
                int s0 = i0, s1 = i1, s2 = i2, tmp;
                if (s0 > s1) { tmp = s0; s0 = s1; s1 = tmp; }
                if (s1 > s2) { tmp = s1; s1 = s2; s2 = tmp; }
                if (s0 > s1) { tmp = s0; s0 = s1; s1 = tmp; }
                const int   sidx[3] = {s0, s1, s2};
                const bool  valid[3] = {s0 < NBODY, s1 < NBODY, s2 < NBODY};

                // ---- temporal-relation feature: tempered softmax then masked softmin ----
                float pvsum = 0.0f, cnum = 0.0f, cden = 0.0f;
                #pragma unroll
                for (int p = 0; p < 3; ++p) {
                    float pv = (valid[PA[p]] && valid[PB[p]]) ? 1.0f : 0.0f;
                    int pc = sidx[PA[p]]; if (pc > NBODY - 1) pc = NBODY - 1;
                    int qc = sidx[PB[p]]; if (qc > NBODY - 1) qc = NBODY - 1;
                    float td = dsr[pc] - dsr[qc];
                    float c0 = (td >  TOLR)        ? 1.0f : 0.0f;
                    float c1 = (fabsf(td) < TOLR)  ? 1.0f : 0.0f;
                    float c2 = (td < -TOLR)        ? 1.0f : 0.0f;
                    float c3 = 1.0f - p0 * c0 - p1 * c1 - p2 * c2;
                    float x0 = c0 * p0, x1 = c1 * p1, x2 = c2 * p2, x3 = c3 * p3;
                    float w0 = expf(x0 / TEMP), w1 = expf(x1 / TEMP);
                    float w2 = expf(x2 / TEMP), w3 = expf(x3 / TEMP);
                    float rrf = (w0 * x0 + w1 * x1 + w2 * x2 + w3 * x3)
                              / (w0 + w1 + w2 + w3);
                    float wgt = pv * expf(-rrf / TEMP);
                    cnum += wgt * rrf; cden += wgt; pvsum += pv;
                }
                float rel = (pvsum > 0.0f) ? (cnum / fmaxf(cden, 1e-30f)) : 1.0f;

                // ---- formula feature from WMMA dot ----
                float dot  = sdot[wv][lane][i];
                float feat = expf(-fabsf(dot - (float)KSEL) / SIGMA) * rel;

                // ---- latest selected body event time ----
                float mbt = 0.0f;
                #pragma unroll
                for (int p = 0; p < NBODY; ++p) {
                    float v = (dsr[p] <= tb) ? dsr[p] * A[i * NCOL + p] : 0.0f;
                    mbt = fmaxf(mbt, v);
                }

                float sg  = 1.0f / (1.0f + expf(-(tb - mbt)));
                float cur = basv + sg * feat * fw[i];
                float lps = logf(cur) - tb * cur
                          + sg * (-mbt * basv + mbt * cur) + logf(pi[i + 1]);
                out[brow * 3 + 1 + i] = lps;
            }
        }
    }
}

extern "C" void kernel_launch(void* const* d_in, const int* in_sizes, int n_in,
                              void* d_out, int out_size, void* d_ws, size_t ws_size,
                              hipStream_t stream) {
    (void)in_sizes; (void)n_in; (void)out_size; (void)d_ws; (void)ws_size;
    const float* t    = (const float*)d_in[0];   // (500,1)
    const float* ds   = (const float*)d_in[1];   // (500,30)
    const float* pi   = (const float*)d_in[2];   // (3,)
    const float* A    = (const float*)d_in[3];   // (2,32)
    const float* base = (const float*)d_in[4];   // (1,)
    const float* fw   = (const float*)d_in[5];   // (2,)
    const float* prob = (const float*)d_in[6];   // (4,)
    float* out = (float*)d_out;                  // (500,3)

    logic_model_fused<<<NBLOCKS, THREADS, 0, stream>>>(t, ds, pi, A, base, fw, prob, out);
}
